// PrunedKVAttentionOptimized_16801912062298
// MI455X (gfx1250) — compile-verified, matrix operand order checked by reading, not run on hardware
//
#include <hip/hip_runtime.h>
#include <math.h>

// ---------------- problem constants ----------------
#define BATCH   2
#define TT      2048
#define CC      1024
#define C3      3072
#define HH      16
#define DD      64
#define BH      32          // BATCH*HH
#define KKEEP   391
#define KPAD    416         // pad to multiple of 32
#define RECENCY 64
#define INV_SQRT_D 0.125f
#define NEG_BIG (-1.0e30f)

#ifndef USE_TDM
#define USE_TDM 1
#endif

typedef __bf16 bf16_t;
typedef __attribute__((ext_vector_type(16))) __bf16 v16bf;
typedef __attribute__((ext_vector_type(8)))  float  v8f;
typedef __attribute__((ext_vector_type(4)))  unsigned int v4u;
typedef __attribute__((ext_vector_type(8)))  int    v8i;
typedef __attribute__((ext_vector_type(4)))  int    v4i;

union BF2 { unsigned int u; struct { __bf16 lo, hi; } s; };
union BF1 { unsigned short u; __bf16 b; };

__device__ __forceinline__ bf16_t f2bf(float x) {
    unsigned int u = __float_as_uint(x);
    u += 0x7fffu + ((u >> 16) & 1u);     // round-to-nearest-even
    BF1 r; r.u = (unsigned short)(u >> 16);
    return r.b;
}

// Load a 16x32 bf16 fragment (A layout; also used for B which reads rows of
// the [N,K] matrix).  CDNA5 16-bit A layout: lanes 0-15 hold K={0..7,16..23},
// lanes 16-31 hold K={8..15,24..31}; bf16 pairs per dword.
__device__ __forceinline__ v16bf load_frag(const bf16_t* row) {
    const int hi = (threadIdx.x >> 4) & 1;
    const unsigned int* p = (const unsigned int*)row + hi * 4;  // hi*8 bf16
    v16bf f;
#pragma unroll
    for (int pr = 0; pr < 8; ++pr) {
        const int dw = (pr < 4) ? pr : (pr + 4);  // k/2: {0,2,4,6,16,18,20,22}
        BF2 d; d.u = p[dw];
        f[2 * pr]     = d.s.lo;
        f[2 * pr + 1] = d.s.hi;
    }
    return f;
}

#define WMMA_BF16(A_, B_, C_) \
    __builtin_amdgcn_wmma_f32_16x16x32_bf16(false, (A_), false, (B_), (short)0, (C_), false, false)

// ---------------- TDM: 2D tensor tile Global -> LDS ------------------------
// Builds a minimal D# (group0 + group1) per CDNA5 ISA 8.3/8.4 and issues
// tensor_load_to_lds.  2D tile: tile_d1 lines of tile_d0 elements (bf16),
// line stride = stride0 elements.  Caller must s_wait_tensorcnt.
__device__ __forceinline__ void tdm_load_2d_bf16(unsigned int lds_off, const void* gptr,
                                                 unsigned int tile_d0, unsigned int tile_d1,
                                                 unsigned long long stride0) {
#if USE_TDM
    unsigned long long ga = (unsigned long long)gptr;
    v4u g0;
    g0[0] = 1u;                                   // count=1, user descriptor
    g0[1] = lds_off;                              // LDS byte address
    g0[2] = (unsigned int)ga;                     // global_addr[31:0]
    g0[3] = (unsigned int)((ga >> 32) & 0x1FFFFFFu) | (2u << 30);  // [56:32] | type=2
    v8i g1;
    g1[0] = (int)(1u << 16);                      // workgroup_mask=0, data_size=1 (2B)
    g1[1] = (int)((tile_d0 & 0xFFFFu) << 16);     // tensor_dim0[15:0]   @ bits 63:48
    g1[2] = (int)(((tile_d0 >> 16) & 0xFFFFu) | ((tile_d1 & 0xFFFFu) << 16)); // td0 hi | td1 lo
    g1[3] = (int)(((tile_d1 >> 16) & 0xFFFFu) | ((tile_d0 & 0xFFFFu) << 16)); // td1 hi | tile_dim0
    g1[4] = (int)(tile_d1 & 0xFFFFu);             // tile_dim1 ; tile_dim2=0
    g1[5] = (int)(unsigned int)(stride0 & 0xFFFFFFFFu);   // tensor_dim0_stride lo
    g1[6] = (int)((stride0 >> 32) & 0xFFFFu);             // stride0 hi ; stride1 lo = 0
    g1[7] = 0;
    v4i z4 = {0, 0, 0, 0};
#if defined(__clang_major__) && (__clang_major__ >= 23)
    v8i z8 = {0, 0, 0, 0, 0, 0, 0, 0};
    __builtin_amdgcn_tensor_load_to_lds(g0, g1, z4, z4, z8, 0);
#else
    __builtin_amdgcn_tensor_load_to_lds(g0, g1, z4, z4, 0);
#endif
#endif
}

// ---------------- elementwise helpers ----------------
__global__ void k_cvt_bf16(const float* __restrict__ s, bf16_t* __restrict__ d, int n) {
    int i = blockIdx.x * blockDim.x + threadIdx.x;
    if (i < n) d[i] = f2bf(s[i]);
}

__global__ void k_zero_f32(float* __restrict__ p, int n) {
    int i = blockIdx.x * blockDim.x + threadIdx.x;
    if (i < n) p[i] = 0.0f;
}

// ---------------- GEMM: C[M,N] = A[M,K] * B[N,K]^T  (bf16 in, f32 acc) ------
// One wave computes a 32x64 strip: 8 accumulators, register double-buffered
// k-loop so next-step fragments are in flight behind the current WMMAs.
__global__ void k_gemm_bf16_nt(const bf16_t* __restrict__ A,
                               const bf16_t* __restrict__ Bm,
                               bf16_t* __restrict__ Cb,
                               float*  __restrict__ Cf,
                               int M, int N, int K) {
    const int wavesPerBlock = blockDim.x >> 5;
    const int wave = blockIdx.x * wavesPerBlock + (threadIdx.x >> 5);
    const int stripsN = N >> 6;
    const int tm = (wave / stripsN) << 5;      // 32 rows
    const int tn = (wave % stripsN) << 6;      // 64 cols
    if (tm >= M) return;

    const int lane = threadIdx.x & 31;
    const int col  = lane & 15;
    const int hi   = lane >> 4;

    v8f acc[2][4] = {};
    const bf16_t* ar0 = A  + (size_t)(tm + col) * K;
    const bf16_t* ar1 = A  + (size_t)(tm + 16 + col) * K;
    const bf16_t* br  = Bm + (size_t)(tn + col) * K;

    v16bf a0 = load_frag(ar0);
    v16bf a1 = load_frag(ar1);
    v16bf b0 = load_frag(br);
    v16bf b1 = load_frag(br + (size_t)16 * K);
    v16bf b2 = load_frag(br + (size_t)32 * K);
    v16bf b3 = load_frag(br + (size_t)48 * K);

    for (int k0 = 0; k0 < K - 32; k0 += 32) {
        const int kn = k0 + 32;
        v16bf na0 = load_frag(ar0 + kn);
        v16bf na1 = load_frag(ar1 + kn);
        v16bf nb0 = load_frag(br + kn);
        v16bf nb1 = load_frag(br + (size_t)16 * K + kn);
        v16bf nb2 = load_frag(br + (size_t)32 * K + kn);
        v16bf nb3 = load_frag(br + (size_t)48 * K + kn);
        __builtin_prefetch(ar0 + kn + 256, 0, 1);   // global_prefetch_b8

        acc[0][0] = WMMA_BF16(a0, b0, acc[0][0]);
        acc[1][0] = WMMA_BF16(a1, b0, acc[1][0]);
        acc[0][1] = WMMA_BF16(a0, b1, acc[0][1]);
        acc[1][1] = WMMA_BF16(a1, b1, acc[1][1]);
        acc[0][2] = WMMA_BF16(a0, b2, acc[0][2]);
        acc[1][2] = WMMA_BF16(a1, b2, acc[1][2]);
        acc[0][3] = WMMA_BF16(a0, b3, acc[0][3]);
        acc[1][3] = WMMA_BF16(a1, b3, acc[1][3]);

        a0 = na0; a1 = na1; b0 = nb0; b1 = nb1; b2 = nb2; b3 = nb3;
    }
    // tail step
    acc[0][0] = WMMA_BF16(a0, b0, acc[0][0]);
    acc[1][0] = WMMA_BF16(a1, b0, acc[1][0]);
    acc[0][1] = WMMA_BF16(a0, b1, acc[0][1]);
    acc[1][1] = WMMA_BF16(a1, b1, acc[1][1]);
    acc[0][2] = WMMA_BF16(a0, b2, acc[0][2]);
    acc[1][2] = WMMA_BF16(a1, b2, acc[1][2]);
    acc[0][3] = WMMA_BF16(a0, b3, acc[0][3]);
    acc[1][3] = WMMA_BF16(a1, b3, acc[1][3]);

#pragma unroll
    for (int t = 0; t < 2; ++t)
#pragma unroll
        for (int s = 0; s < 4; ++s)
#pragma unroll
            for (int r = 0; r < 8; ++r) {
                size_t off = (size_t)(tm + t * 16 + r + hi * 8) * N + tn + s * 16 + col;
                if (Cf) Cf[off] = acc[t][s][r];
                else    Cb[off] = f2bf(acc[t][s][r]);
            }
}

// ---------------- importance: mean over queries of causal softmax ----------
// grid.x = BH*16 ; block = 256 (8 waves).  Each wave owns one 16-query tile.
__global__ void k_importance(const bf16_t* __restrict__ qkv, float* __restrict__ imp) {
    const int bh = blockIdx.x >> 4;
    const int qg = blockIdx.x & 15;
    const int b  = bh >> 4;
    const int h  = bh & 15;
    const int w  = threadIdx.x >> 5;
    const int lane = threadIdx.x & 31;
    const int col  = lane & 15;
    const int hi   = lane >> 4;
    const int qt   = qg * 8 + w;          // 0..127
    const int q0   = qt << 4;

    const bf16_t* qrow = qkv + (size_t)(b * TT + q0 + col) * C3 + h * DD;
    const v16bf aq0 = load_frag(qrow);        // Q fragments: loop-invariant
    const v16bf aq1 = load_frag(qrow + 32);

    float m[8], l[8];
#pragma unroll
    for (int r = 0; r < 8; ++r) { m[r] = NEG_BIG; l[r] = 0.0f; }

    // ---- pass 1: online row max / sum-exp over causal key tiles ----
    for (int kt = 0; kt <= qt; ++kt) {
        const bf16_t* krow = qkv + (size_t)(b * TT + (kt << 4) + col) * C3 + CC + h * DD;
        v8f s = {};
        s = WMMA_BF16(aq0, load_frag(krow), s);
        s = WMMA_BF16(aq1, load_frag(krow + 32), s);
        const int kcol = (kt << 4) + col;
#pragma unroll
        for (int r = 0; r < 8; ++r) {
            const int qrow_g = q0 + r + hi * 8;
            float sv = s[r] * INV_SQRT_D;
            sv = (kcol <= qrow_g) ? sv : NEG_BIG;
            float mx = sv;
#pragma unroll
            for (int off = 1; off < 16; off <<= 1) mx = fmaxf(mx, __shfl_xor(mx, off));
            float mn = fmaxf(m[r], mx);
            float sc = __expf(m[r] - mn);
            float p  = __expf(sv - mn);
            float rs = p;
#pragma unroll
            for (int off = 1; off < 16; off <<= 1) rs += __shfl_xor(rs, off);
            l[r] = l[r] * sc + rs;
            m[r] = mn;
        }
    }

    // ---- pass 2: normalized column sums -> importance atomics ----
    for (int kt = 0; kt <= qt; ++kt) {
        const bf16_t* krow = qkv + (size_t)(b * TT + (kt << 4) + col) * C3 + CC + h * DD;
        v8f s = {};
        s = WMMA_BF16(aq0, load_frag(krow), s);
        s = WMMA_BF16(aq1, load_frag(krow + 32), s);
        const int kcol = (kt << 4) + col;
        float colp = 0.0f;
#pragma unroll
        for (int r = 0; r < 8; ++r) {
            const int qrow_g = q0 + r + hi * 8;
            float sv = s[r] * INV_SQRT_D;
            sv = (kcol <= qrow_g) ? sv : NEG_BIG;
            colp += __expf(sv - m[r]) / l[r];
        }
        colp += __shfl_xor(colp, 16);     // combine both halves (rows 0-7 + 8-15)
        if (hi == 0)
            atomicAdd(&imp[(size_t)bh * TT + kcol], colp * (1.0f / (float)TT));
    }
}

// ---------------- top-k select per (b,h): binary search on float bits ------
__global__ void k_topk(const float* __restrict__ imp, int* __restrict__ idx_out) {
    const int bh  = blockIdx.x;
    const int tid = threadIdx.x;
    __shared__ float vals[TT];
    __shared__ int cnt;
    __shared__ int pos;

    for (int i = tid; i < TT; i += 256) {
        float x = imp[(size_t)bh * TT + i];
        if (i >= TT - RECENCY) x = 1.0f;   // recency override
        vals[i] = x;
    }
    __syncthreads();

    unsigned int lo = 0u, hiu = 0x40000000u;   // [0, 2.0f]
    for (int it = 0; it < 30; ++it) {
        unsigned int mid = lo + ((hiu - lo) >> 1);
        float mf = __uint_as_float(mid);
        if (tid == 0) cnt = 0;
        __syncthreads();
        int c = 0;
        for (int i = tid; i < TT; i += 256) c += (vals[i] > mf);
        atomicAdd(&cnt, c);
        __syncthreads();
        int total = cnt;
        __syncthreads();
        if (total > KKEEP) lo = mid; else hiu = mid;
    }
    const float thr = __uint_as_float(hiu);   // count(> thr) <= KKEEP

    if (tid == 0) pos = 0;
    __syncthreads();
    for (int i = tid; i < TT; i += 256) {
        if (vals[i] > thr) {
            int p = atomicAdd(&pos, 1);
            if (p < KKEEP) idx_out[bh * KKEEP + p] = i;
        }
    }
    __syncthreads();
    for (int i = tid; i < TT; i += 256) {
        if (vals[i] <= thr) {
            int p = atomicAdd(&pos, 1);
            if (p < KKEEP) idx_out[bh * KKEEP + p] = i;
        }
    }
}

// ---------------- gather pruned K (row-major) and V (transposed) -----------
__global__ void k_gather(const bf16_t* __restrict__ qkv, const int* __restrict__ idx,
                         bf16_t* __restrict__ kk, bf16_t* __restrict__ vt) {
    int i = blockIdx.x * blockDim.x + threadIdx.x;   // over BH*KPAD*DD
    if (i >= BH * KPAD * DD) return;
    const int d  = i & (DD - 1);
    const int kr = (i >> 6) % KPAD;
    const int bh = i / (KPAD * DD);
    const int b  = bh >> 4;
    const int h  = bh & 15;
    BF1 z; z.u = 0;
    bf16_t kval = z.b, vval = z.b;
    if (kr < KKEEP) {
        int t = idx[bh * KKEEP + kr];
        size_t base = (size_t)(b * TT + t) * C3 + h * DD + d;
        kval = qkv[base + CC];
        vval = qkv[base + 2 * CC];
    }
    kk[((size_t)bh * KPAD + kr) * DD + d] = kval;         // [bh][kr][d]
    vt[((size_t)bh * DD + d) * KPAD + kr] = vval;         // [bh][d][kr]
}

// ---------------- second (pruned) attention, flash-style, WMMA -------------
// grid.x = BH*16, block = 256 (8 waves).  All waves in a block share (b,h):
// wave 0 DMA's the shared 32-key K/V tiles into LDS via the Tensor Data
// Mover each step; waves read fragments from LDS.
__global__ void k_attn2(const bf16_t* __restrict__ qkv,
                        const bf16_t* __restrict__ kk,
                        const bf16_t* __restrict__ vt,
                        bf16_t* __restrict__ outb) {
    __shared__ __align__(16) __bf16 kt_lds[32 * DD];     // [key][d]  4 KB
    __shared__ __align__(16) __bf16 vt_lds[DD * 32];     // [d][key]  4 KB
    __shared__ __align__(16) __bf16 pbuf[8][16 * 32];    // wave-private P

    const int bh = blockIdx.x >> 4;
    const int qg = blockIdx.x & 15;
    const int b  = bh >> 4;
    const int h  = bh & 15;
    const int w  = threadIdx.x >> 5;
    const int lane = threadIdx.x & 31;
    const int col  = lane & 15;
    const int hi   = lane >> 4;
    const int q0   = (qg * 8 + w) << 4;

    const bf16_t* qrow = qkv + (size_t)(b * TT + q0 + col) * C3 + h * DD;
    const bf16_t* kbh  = kk + (size_t)bh * KPAD * DD;
    const bf16_t* vbh  = vt + (size_t)bh * DD * KPAD;

    const v16bf aq0 = load_frag(qrow);        // Q fragments: loop-invariant
    const v16bf aq1 = load_frag(qrow + 32);

    const unsigned int lds_k = (unsigned int)(unsigned long long)(void*)kt_lds;
    const unsigned int lds_v = (unsigned int)(unsigned long long)(void*)vt_lds;

    float m[8], l[8];
#pragma unroll
    for (int r = 0; r < 8; ++r) { m[r] = NEG_BIG; l[r] = 0.0f; }
    v8f o[4] = {};

    for (int kt2 = 0; kt2 < KPAD / 32; ++kt2) {
        const int kbase = kt2 * 32;
        __syncthreads();                       // previous tile fully consumed
        if (w == 0) {
            // K tile: 32 contiguous rows of 64 bf16
            tdm_load_2d_bf16(lds_k, kbh + (size_t)kbase * DD, DD, 32, DD);
            // V tile: 64 rows of 32 bf16, row stride KPAD
            tdm_load_2d_bf16(lds_v, vbh + kbase, 32, DD, KPAD);
            __builtin_amdgcn_s_wait_tensorcnt(0);
        }
        __syncthreads();                       // tile ready

        const bf16_t* kr0 = kt_lds + (size_t)col * DD;
        const bf16_t* kr1 = kt_lds + (size_t)(16 + col) * DD;
        v8f s0 = {}, s1 = {};
        s0 = WMMA_BF16(aq0, load_frag(kr0), s0);
        s0 = WMMA_BF16(aq1, load_frag(kr0 + 32), s0);
        s1 = WMMA_BF16(aq0, load_frag(kr1), s1);
        s1 = WMMA_BF16(aq1, load_frag(kr1 + 32), s1);

        const int key0 = kbase + col;
        const int key1 = kbase + 16 + col;
        float sc[8];
#pragma unroll
        for (int r = 0; r < 8; ++r) {
            float v0 = (key0 < KKEEP) ? s0[r] * INV_SQRT_D : NEG_BIG;
            float v1 = (key1 < KKEEP) ? s1[r] * INV_SQRT_D : NEG_BIG;
            float mx = fmaxf(v0, v1);
#pragma unroll
            for (int off = 1; off < 16; off <<= 1) mx = fmaxf(mx, __shfl_xor(mx, off));
            float mn = fmaxf(m[r], mx);
            sc[r] = __expf(m[r] - mn);
            float p0 = __expf(v0 - mn);
            float p1 = __expf(v1 - mn);
            float rs = p0 + p1;
#pragma unroll
            for (int off = 1; off < 16; off <<= 1) rs += __shfl_xor(rs, off);
            l[r] = l[r] * sc[r] + rs;
            m[r] = mn;
            const int row = r + hi * 8;        // stage P (row=query, col=key)
            pbuf[w][row * 32 + col]      = f2bf(p0);
            pbuf[w][row * 32 + 16 + col] = f2bf(p1);
        }
#pragma unroll
        for (int j = 0; j < 4; ++j)            // rescale running output
#pragma unroll
            for (int r = 0; r < 8; ++r) o[j][r] *= sc[r];

        // O += P * V  (A = P from LDS, B rows are d, K-dim = 32 keys)
        v16bf ap = load_frag(&pbuf[w][col * 32]);
#pragma unroll
        for (int j = 0; j < 4; ++j) {
            const bf16_t* vrow = vt_lds + (size_t)(j * 16 + col) * 32;
            o[j] = WMMA_BF16(ap, load_frag(vrow), o[j]);
        }
    }

#pragma unroll
    for (int j = 0; j < 4; ++j)
#pragma unroll
        for (int r = 0; r < 8; ++r) {
            const int t = q0 + r + hi * 8;
            outb[(size_t)(b * TT + t) * CC + h * DD + j * 16 + col] = f2bf(o[j][r] / l[r]);
        }
}

// ---------------- launcher -------------------------------------------------
extern "C" void kernel_launch(void* const* d_in, const int* in_sizes, int n_in,
                              void* d_out, int out_size, void* d_ws, size_t ws_size,
                              hipStream_t stream) {
    const float* x      = (const float*)d_in[0];   // [B,T,C]
    const float* w_attn = (const float*)d_in[1];   // [3C,C]
    const float* w_proj = (const float*)d_in[2];   // [C,C]
    float* out = (float*)d_out;                    // [B,T,C]

    char* ws = (char*)d_ws;
    size_t off = 0;
    auto alloc = [&](size_t bytes) { void* p = ws + off; off = (off + bytes + 255) & ~(size_t)255; return p; };

    bf16_t* x_bf   = (bf16_t*)alloc((size_t)BATCH * TT * CC * 2);
    bf16_t* wa_bf  = (bf16_t*)alloc((size_t)C3 * CC * 2);
    bf16_t* wp_bf  = (bf16_t*)alloc((size_t)CC * CC * 2);
    bf16_t* qkv_bf = (bf16_t*)alloc((size_t)BATCH * TT * C3 * 2);
    float*  imp    = (float*) alloc((size_t)BH * TT * 4);
    int*    idx    = (int*)   alloc((size_t)BH * KKEEP * 4);
    bf16_t* kk     = (bf16_t*)alloc((size_t)BH * KPAD * DD * 2);
    bf16_t* vt     = (bf16_t*)alloc((size_t)BH * KPAD * DD * 2);
    bf16_t* aout   = (bf16_t*)alloc((size_t)BATCH * TT * CC * 2);

    const int nx  = BATCH * TT * CC;   // 4,194,304
    const int nwa = C3 * CC;           // 3,145,728
    const int nwp = CC * CC;           // 1,048,576

    k_cvt_bf16<<<(nx  + 255) / 256, 256, 0, stream>>>(x, x_bf, nx);
    k_cvt_bf16<<<(nwa + 255) / 256, 256, 0, stream>>>(w_attn, wa_bf, nwa);
    k_cvt_bf16<<<(nwp + 255) / 256, 256, 0, stream>>>(w_proj, wp_bf, nwp);

    // qkv = x @ w_attn^T : M=4096, N=3072, K=1024 ; waves = (M/32)*(N/64)
    {
        const int M = BATCH * TT, N = C3, K = CC;
        const int waves = (M / 32) * (N / 64);
        k_gemm_bf16_nt<<<waves / 8, 256, 0, stream>>>(x_bf, wa_bf, qkv_bf, (float*)nullptr, M, N, K);
    }

    k_zero_f32<<<(BH * TT + 255) / 256, 256, 0, stream>>>(imp, BH * TT);
    k_importance<<<BH * 16, 256, 0, stream>>>(qkv_bf, imp);
    k_topk<<<BH, 256, 0, stream>>>(imp, idx);

    {
        const int n = BH * KPAD * DD;
        k_gather<<<(n + 255) / 256, 256, 0, stream>>>(qkv_bf, idx, kk, vt);
    }

    k_attn2<<<BH * 16, 256, 0, stream>>>(qkv_bf, kk, vt, aout);

    // out = aout @ w_proj^T : M=4096, N=1024, K=1024
    {
        const int M = BATCH * TT, N = CC, K = CC;
        const int waves = (M / 32) * (N / 64);
        k_gemm_bf16_nt<<<waves / 8, 256, 0, stream>>>(aout, wp_bf, (bf16_t*)nullptr, out, M, N, K);
    }
}
